// Net_latency_49520972923432
// MI455X (gfx1250) — matrix operand order repaired and checked by reading, not compiled, not run on hardware
//
#include <hip/hip_runtime.h>
#include <math.h>

// ---------------- types for WMMA ----------------
typedef __attribute__((ext_vector_type(16))) __bf16          v16bf;
typedef __attribute__((ext_vector_type(8)))  float           v8f;
typedef __attribute__((ext_vector_type(8)))  unsigned short  v8u16;

struct U16x16 { v8u16 lo, hi; };   // 32 bytes, bit-castable to v16bf

__device__ __forceinline__ unsigned short f2bf(float f) {
    unsigned u = __builtin_bit_cast(unsigned, f);
    u += 0x7fffu + ((u >> 16) & 1u);     // round-to-nearest-even
    return (unsigned short)(u >> 16);
}
__device__ __forceinline__ float bf2f(unsigned short u) {
    return __builtin_bit_cast(float, (unsigned)u << 16);
}

static constexpr int  BB  = 1024;   // batch
static constexpr int  NN  = 128;    // nodes per tree

// ---------------- prep kernels ----------------
__global__ void zero_u32(unsigned int* p, size_t n) {
    size_t i = (size_t)blockIdx.x * blockDim.x + threadIdx.x;
    size_t stride = (size_t)gridDim.x * blockDim.x;
    for (; i < n; i += stride) p[i] = 0u;
}

// trees fp32 [B][318][128] -> bf16 [B][128][320]  (channels 318..319 = 0)
__global__ void transpose_trees(const float* __restrict__ in, unsigned short* __restrict__ out) {
    size_t tid = (size_t)blockIdx.x * blockDim.x + threadIdx.x;
    const size_t total = (size_t)BB * NN * 320;
    if (tid >= total) return;
    int c = (int)(tid % 320);
    size_t r = tid / 320;
    int n = (int)(r % NN);
    int b = (int)(r / NN);
    unsigned short v = 0;
    if (c < 318) v = f2bf(in[((size_t)b * 318 + c) * NN + n]);
    out[tid] = v;
}

// enc_w fp32 [109][318] -> bf16 [128][320] zero-padded (rows 109..127 zero)
__global__ void pack_wenc(const float* __restrict__ w, unsigned short* __restrict__ out) {
    int tid = blockIdx.x * blockDim.x + threadIdx.x;
    const int total = 128 * 320;
    if (tid >= total) return;
    int c = tid % 320, o = tid / 320;
    unsigned short v = 0;
    if (o < 109 && c < 318) v = f2bf(w[o * 318 + c]);
    out[tid] = v;
}

__global__ void pack_benc(const float* __restrict__ b, float* __restrict__ out) {
    int o = blockIdx.x * blockDim.x + threadIdx.x;
    if (o >= 128) return;
    out[o] = (o < 109) ? b[o] : 0.0f;
}

// conv w fp32 [Cout][Cin][3] -> bf16 [Cout][3*Cinp], kk = j*Cinp + c, zero pad c>=Cin
__global__ void pack_wconv(const float* __restrict__ w, unsigned short* __restrict__ out,
                           int Cin, int Cinp, int Cout) {
    int tid = blockIdx.x * blockDim.x + threadIdx.x;
    int total = Cout * 3 * Cinp;
    if (tid >= total) return;
    int c = tid % Cinp;
    int j = (tid / Cinp) % 3;
    int o = tid / (3 * Cinp);
    unsigned short v = 0;
    if (c < Cin) v = f2bf(w[((size_t)o * Cin + c) * 3 + j]);
    out[tid] = v;
}

// ---------------- WMMA GEMM (optionally fused with tree gather) ----------------
// D[o,node] = sum_k W[o,k] * Bmat[k,node] + bias[o]
//   W    : bf16 [Cout][Kp]           (prepacked, K order = j*Cinp + c; Cout % 64 == 0)
//   act  : bf16 [B][N][Cinp]         (node-major, channels contiguous, Cinp % 32 == 0)
//   GATHER: B-columns come from rows idx[b][3n+j]; Kp == 3*Cinp
// one wave per block; wave computes 64(O) x 16(node) via 4 accumulators.
// K-loop: ping-pong double buffer, unrolled x2 (step count is always even);
// the trailing prefetch overfetches one 32-elem step (stays inside workspace).
template <bool GATHER, bool STORE_BF16>
__global__ __launch_bounds__(32)
void tree_gemm(const unsigned short* __restrict__ W,
               const unsigned short* __restrict__ act,
               const int* __restrict__ idx,
               const float* __restrict__ bias,
               float* __restrict__ yout,
               unsigned short* __restrict__ aout,
               int Cinp, int Kp, int Cstore) {
    constexpr int OT = 4;
    const int lane = threadIdx.x;
    const int lm = lane & 15;
    const int hs = lane >> 4;
    const int og = blockIdx.x;
    const int ntile = blockIdx.y;
    const int b = blockIdx.z;
    const int node = ntile * 16 + lm;

    const unsigned short* rowp[3];
    if (GATHER) {
        const int* ip = idx + (size_t)b * 3 * NN + 3 * node;
        const unsigned short* base = act + (size_t)b * NN * Cinp;
        rowp[0] = base + (size_t)ip[0] * Cinp;
        rowp[1] = base + (size_t)ip[1] * Cinp;
        rowp[2] = base + (size_t)ip[2] * Cinp;
    } else {
        rowp[0] = act + ((size_t)b * NN + node) * Cinp;
    }

    v8f acc[OT];
#pragma unroll
    for (int t = 0; t < OT; ++t)
#pragma unroll
        for (int e = 0; e < 8; ++e) acc[t][e] = 0.0f;

    const int nsteps = (GATHER ? 3 : 1) * (Cinp / 32);   // always even here

    // rolling pointers
    // A: row M = lm of o-tile t lives at W[(og*64 + t*16 + lm)*Kp]; lane chunks at +hs*8 and +16+hs*8
    const unsigned short* wptr = W + ((size_t)og * OT * 16 + lm) * Kp + hs * 8;
    // B: column `node`, lane chunks at +hs*8 and +16+hs*8
    const unsigned short* bptr = rowp[0] + hs * 8;
    int kc = 0, j = 0;

    auto loadFrag = [&](U16x16& bdst, U16x16 (&adst)[OT]) {
        bdst.lo = *(const v8u16*)(bptr);
        bdst.hi = *(const v8u16*)(bptr + 16);
#pragma unroll
        for (int t = 0; t < OT; ++t) {
            const unsigned short* wp = wptr + (size_t)t * 16 * Kp;
            adst[t].lo = *(const v8u16*)(wp);
            adst[t].hi = *(const v8u16*)(wp + 16);
        }
    };
    auto advance = [&]() {
        wptr += 32;
        if (GATHER) {
            kc += 32;
            if (kc == Cinp) { kc = 0; j = (j < 2) ? j + 1 : 0; bptr = rowp[j] + hs * 8; }
            else            { bptr += 32; }
        } else {
            bptr += 32;
        }
    };
    auto mmac = [&](U16x16& bsrc, U16x16 (&asrc)[OT]) {
        v16bf bf = __builtin_bit_cast(v16bf, bsrc);
#pragma unroll
        for (int t = 0; t < OT; ++t) {
            v16bf af = __builtin_bit_cast(v16bf, asrc[t]);
            acc[t] = __builtin_amdgcn_wmma_f32_16x16x32_bf16(
                false, af, false, bf, (short)0, acc[t], false, false);
        }
    };

    U16x16 buA, auA[OT], buB, auB[OT];
    loadFrag(buA, auA);                       // step 0
    for (int s = 0; s < nsteps; s += 2) {
        advance();
        loadFrag(buB, auB);                   // step s+1
        mmac(buA, auA);                       // consume step s (other set in flight)
        advance();
        loadFrag(buA, auA);                   // step s+2 (overfetch on last iter)
        mmac(buB, auB);                       // consume step s+1
    }

    // ---- LDS transpose for coalesced stores ----
    // D layout: VGPR r -> M = t*16 + 8*hs + r, N(row/node) = lm
    __shared__ float lds[16][65];
#pragma unroll
    for (int t = 0; t < OT; ++t)
#pragma unroll
        for (int r = 0; r < 8; ++r) {
            int o = t * 16 + hs * 8 + r;
            lds[lm][o] = acc[t][r] + bias[og * 64 + o];
        }
    __syncthreads();

    const int row  = lane >> 1;         // node within tile
    const int col0 = (lane & 1) * 32;   // o within 64-group
    const size_t dbase = ((size_t)b * NN + ntile * 16 + row) * Cstore + og * 64 + col0;
    if (STORE_BF16) {
        unsigned short* dst = aout + dbase;
#pragma unroll
        for (int e = 0; e < 32; e += 2) {
            unsigned v = (unsigned)f2bf(lds[row][col0 + e]) |
                         ((unsigned)f2bf(lds[row][col0 + e + 1]) << 16);
            *(unsigned*)(dst + e) = v;
        }
    } else {
        float* dst = yout + dbase;
#pragma unroll
        for (int e = 0; e < 32; e += 4) {
            float4 v = make_float4(lds[row][col0 + e],     lds[row][col0 + e + 1],
                                   lds[row][col0 + e + 2], lds[row][col0 + e + 3]);
            *(float4*)(dst + e) = v;
        }
    }
}

// ---------------- global LayerNorm: reduction / finalize / apply ----------------
__global__ void reduce_sums(const float4* __restrict__ y, size_t n4, double* __restrict__ out) {
    double s = 0.0, q = 0.0;
    size_t i = (size_t)blockIdx.x * blockDim.x + threadIdx.x;
    size_t stride = (size_t)gridDim.x * blockDim.x;
    for (; i < n4; i += stride) {
        float4 v = y[i];
        s += (double)v.x + (double)v.y + (double)v.z + (double)v.w;
        q += (double)v.x * v.x + (double)v.y * v.y + (double)v.z * v.z + (double)v.w * v.w;
    }
    __shared__ double sh[256], sh2[256];
    int t = threadIdx.x;
    sh[t] = s; sh2[t] = q;
    __syncthreads();
    for (int st = 128; st > 0; st >>= 1) {
        if (t < st) { sh[t] += sh[t + st]; sh2[t] += sh2[t + st]; }
        __syncthreads();
    }
    if (t == 0) {
        atomicAdd(&out[0], sh[0]);
        atomicAdd(&out[1], sh2[0]);
    }
}

__global__ void finalize_stats(const double* __restrict__ s, double cnt, float* __restrict__ muinv) {
    double sum = s[0], ss = s[1];
    double mu = sum / cnt;
    double var = (ss - sum * mu) / (cnt - 1.0);
    double sd = sqrt(var > 0.0 ? var : 0.0);
    muinv[0] = (float)mu;
    muinv[1] = (float)(1.0 / (sd + 1e-5));
}

__global__ void norm_relu(const float* __restrict__ y, const float* __restrict__ muinv,
                          size_t n, unsigned short* __restrict__ out) {
    float mu = muinv[0], inv = muinv[1];
    size_t i = (size_t)blockIdx.x * blockDim.x + threadIdx.x;
    size_t stride = (size_t)gridDim.x * blockDim.x;
    for (; i < n; i += stride)
        out[i] = f2bf(fmaxf(0.0f, (y[i] - mu) * inv));
}

// ---------------- max-pool + heads ----------------
__global__ void final_heads(const unsigned short* __restrict__ act,
                            const float* __restrict__ lw, const float* __restrict__ lb,
                            const float* __restrict__ cw, const float* __restrict__ cb,
                            float* __restrict__ out) {
    int b = blockIdx.x;
    int o = threadIdx.x; // 128
    const unsigned short* p = act + (size_t)b * NN * 128 + o;
    float m = -1e30f;
    for (int n = 0; n < NN; ++n) m = fmaxf(m, bf2f(p[(size_t)n * 128]));
    __shared__ float sh[128];
    sh[o] = m * lw[o];
    __syncthreads();
    for (int s = 64; s > 0; s >>= 1) { if (o < s) sh[o] += sh[o + s]; __syncthreads(); }
    float lat = 0.0f;
    if (o == 0) lat = sh[0];
    __syncthreads();
    sh[o] = m * cw[o];
    __syncthreads();
    for (int s = 64; s > 0; s >>= 1) { if (o < s) sh[o] += sh[o + s]; __syncthreads(); }
    if (o == 0) {
        out[b]      = 1.0f / (1.0f + expf(-(lat   + lb[0])));
        out[BB + b] = 1.0f / (1.0f + expf(-(sh[0] + cb[0])));
    }
}

// ---------------- launcher ----------------
extern "C" void kernel_launch(void* const* d_in, const int* in_sizes, int n_in,
                              void* d_out, int out_size, void* d_ws, size_t ws_size,
                              hipStream_t stream) {
    (void)in_sizes; (void)n_in; (void)out_size; (void)ws_size;
    const float* trees = (const float*)d_in[0];
    const int*   idx   = (const int*)  d_in[1];
    const float* enc_w = (const float*)d_in[2];
    const float* enc_b = (const float*)d_in[3];
    const float* w1 = (const float*)d_in[4];  const float* b1 = (const float*)d_in[5];
    const float* w2 = (const float*)d_in[6];  const float* b2 = (const float*)d_in[7];
    const float* w3 = (const float*)d_in[8];  const float* b3 = (const float*)d_in[9];
    const float* lw = (const float*)d_in[10]; const float* lb = (const float*)d_in[11];
    const float* cw = (const float*)d_in[12]; const float* cb = (const float*)d_in[13];
    float* out = (float*)d_out;

    char* ws = (char*)d_ws;
    size_t off = 0;
    auto carve = [&](size_t bytes) { char* p = ws + off; off = (off + bytes + 255) & ~(size_t)255; return p; };
    unsigned short* tbf   = (unsigned short*)carve((size_t)BB * NN * 320 * 2); // trees bf16
    unsigned short* actA  = (unsigned short*)carve((size_t)BB * NN * 256 * 2); // act1(128)/act3(256)
    unsigned short* actB  = (unsigned short*)carve((size_t)BB * NN * 512 * 2); // act2(512)/act4(128)
    float*          y     = (float*)         carve((size_t)BB * NN * 512 * 4); // conv out fp32
    unsigned short* wenc  = (unsigned short*)carve((size_t)128 * 320 * 2);
    unsigned short* w1p   = (unsigned short*)carve((size_t)512 * 384 * 2);
    unsigned short* w2p   = (unsigned short*)carve((size_t)256 * 1536 * 2);
    unsigned short* w3p   = (unsigned short*)carve((size_t)128 * 768 * 2);
    float*          benc  = (float*)         carve(128 * 4);
    double*         stats = (double*)        carve(6 * 8);
    float*          muinv = (float*)         carve(6 * 4);
    (void)carve(256); // tail pad so any final-step overfetch stays inside d_ws

    // zero stats (6 doubles)
    zero_u32<<<1, 64, 0, stream>>>((unsigned int*)stats, 12);

    // prep
    {
        size_t tot = (size_t)BB * NN * 320;
        transpose_trees<<<(unsigned)((tot + 255) / 256), 256, 0, stream>>>(trees, tbf);
    }
    pack_wenc<<<(128 * 320 + 255) / 256, 256, 0, stream>>>(enc_w, wenc);
    pack_benc<<<1, 128, 0, stream>>>(enc_b, benc);
    pack_wconv<<<(512 * 3 * 128 + 255) / 256, 256, 0, stream>>>(w1, w1p, 109, 128, 512);
    pack_wconv<<<(256 * 3 * 512 + 255) / 256, 256, 0, stream>>>(w2, w2p, 512, 512, 256);
    pack_wconv<<<(128 * 3 * 256 + 255) / 256, 256, 0, stream>>>(w3, w3p, 256, 256, 128);

    // encoder: [128 x 320] * trees -> act1 bf16 [B][N][128] (ch 109..127 exact zeros)
    tree_gemm<false, true><<<dim3(2, NN / 16, BB), 32, 0, stream>>>(
        wenc, tbf, nullptr, benc, nullptr, actA, 320, 320, 128);

    // ---- layer 1: 109(128) -> 512 ----
    tree_gemm<true, false><<<dim3(512 / 64, NN / 16, BB), 32, 0, stream>>>(
        w1p, actA, idx, b1, y, nullptr, 128, 384, 512);
    {
        size_t cnt = (size_t)BB * NN * 512;
        reduce_sums<<<2048, 256, 0, stream>>>((const float4*)y, cnt / 4, stats + 0);
        finalize_stats<<<1, 1, 0, stream>>>(stats + 0, (double)cnt, muinv + 0);
        norm_relu<<<4096, 256, 0, stream>>>(y, muinv + 0, cnt, actB);
    }

    // ---- layer 2: 512 -> 256 ----
    tree_gemm<true, false><<<dim3(256 / 64, NN / 16, BB), 32, 0, stream>>>(
        w2p, actB, idx, b2, y, nullptr, 512, 1536, 256);
    {
        size_t cnt = (size_t)BB * NN * 256;
        reduce_sums<<<2048, 256, 0, stream>>>((const float4*)y, cnt / 4, stats + 2);
        finalize_stats<<<1, 1, 0, stream>>>(stats + 2, (double)cnt, muinv + 2);
        norm_relu<<<4096, 256, 0, stream>>>(y, muinv + 2, cnt, actA);
    }

    // ---- layer 3: 256 -> 128 ----
    tree_gemm<true, false><<<dim3(128 / 64, NN / 16, BB), 32, 0, stream>>>(
        w3p, actA, idx, b3, y, nullptr, 256, 768, 128);
    {
        size_t cnt = (size_t)BB * NN * 128;
        reduce_sums<<<2048, 256, 0, stream>>>((const float4*)y, cnt / 4, stats + 4);
        finalize_stats<<<1, 1, 0, stream>>>(stats + 4, (double)cnt, muinv + 4);
        norm_relu<<<4096, 256, 0, stream>>>(y, muinv + 4, cnt, actB);
    }

    // ---- pool + heads ----
    final_heads<<<BB, 128, 0, stream>>>(actB, lw, lb, cw, cb, out);
}
